// GCN_75385265979507
// MI455X (gfx1250) — matrix operand order repaired and checked by reading, not compiled, not run on hardware
//
#include <hip/hip_runtime.h>
#include <hip/hip_bf16.h>

// ---------------- problem constants ----------------
#define NN   32768   // nodes
#define BB   128     // graphs
#define SS   256     // nodes per graph
#define EE   524288  // edges
#define DIN  128
#define DH   256
#define HH   8
#define HD   32      // DH/H
#define DFF  1024
#define EPS  1e-5f

// ---------------- bf16 / WMMA types ----------------
typedef __bf16 bf16;
typedef __attribute__((ext_vector_type(16))) bf16  v16bf;
typedef __attribute__((ext_vector_type(8)))  float v8f;
typedef __attribute__((ext_vector_type(4)))  unsigned int uint4a;

union AFragU { v16bf v; uint4a q[2]; };

__device__ __forceinline__ bf16 f2bf(float f) {
  // round-to-nearest-even f32 -> bf16 via integer math
  unsigned u = __float_as_uint(f);
  unsigned r = (u + 0x7FFFu + ((u >> 16) & 1u)) >> 16;
  unsigned short s = (unsigned short)r;
  bf16 o; __builtin_memcpy(&o, &s, 2);
  return o;
}

// ---------------- weight convert + transpose (K x N f32 -> N x K bf16) ---------
__global__ void wconv_kernel(const float* __restrict__ w, bf16* __restrict__ wT,
                             int K, int N) {
  int i = blockIdx.x * blockDim.x + threadIdx.x;
  if (i >= K * N) return;
  int k = i / N, n = i - k * N;
  wT[(size_t)n * K + k] = f2bf(w[i]);
}

// ---------------- BN statistics ----------------
// blockDim == C; each block sums a 256-row chunk, coalesced, atomics into sums.
__global__ void colsum_kernel(const float* __restrict__ x, float* __restrict__ sums,
                              float* __restrict__ sumsq, int C, int rowsPerBlock) {
  int f = threadIdx.x;
  size_t r0 = (size_t)blockIdx.x * rowsPerBlock;
  float s = 0.f, q = 0.f;
  for (int r = 0; r < rowsPerBlock; ++r) {
    float v = x[(r0 + r) * C + f];
    s += v; q += v * v;
  }
  atomicAdd(&sums[f], s);
  atomicAdd(&sumsq[f], q);
}

__global__ void bnstats_kernel(const float* __restrict__ sums, const float* __restrict__ sumsq,
                               float* __restrict__ mean, float* __restrict__ invstd,
                               int C, float invN) {
  int f = blockIdx.x * blockDim.x + threadIdx.x;
  if (f >= C) return;
  float mu = sums[f] * invN;
  float var = sumsq[f] * invN - mu * mu;
  mean[f] = mu;
  invstd[f] = rsqrtf(var + EPS);
}

__global__ void bn_apply_kernel(const float* __restrict__ x, const float* __restrict__ mean,
                                const float* __restrict__ invstd, const float* __restrict__ g,
                                const float* __restrict__ b, bf16* __restrict__ out,
                                int total, int C) {
  int i = blockIdx.x * blockDim.x + threadIdx.x;
  if (i >= total) return;
  int f = i % C;
  out[i] = f2bf((x[i] - mean[f]) * invstd[f] * g[f] + b[f]);
}

// ---------------- generic bf16 WMMA GEMM ----------------
// C[M,N] = A[M,K](bf16,row) x BT[N,K](bf16, pre-transposed) (+bias) (+relu)
// wave tile 16x64 (4 accumulators), block = 8 waves -> 128x64.
template<bool OUT_BF16, bool RELU>
__global__ void __launch_bounds__(256)
gemm_bf16_kernel(const bf16* __restrict__ A, const bf16* __restrict__ BT,
                 const float* __restrict__ bias, void* __restrict__ Cout,
                 int M, int N, int K) {
  const int lane = threadIdx.x & 31;
  const int wave = threadIdx.x >> 5;
  const int m16  = lane & 15;
  const int kh   = lane >> 4;
  const int rowBase = blockIdx.y * 128 + wave * 16;
  const int colBase = blockIdx.x * 64;

  v8f acc[4] = {};
  const bf16* aRow = A + (size_t)(rowBase + m16) * K + kh * 16;

  for (int k0 = 0; k0 < K; k0 += 32) {
    __builtin_prefetch(aRow + k0 + 128, 0, 0);   // global_prefetch_b8
    AFragU a;
    a.q[0] = *(const uint4a*)(aRow + k0);        // K = kh*16 + 0..7
    a.q[1] = *(const uint4a*)(aRow + k0 + 8);    // K = kh*16 + 8..15
#pragma unroll
    for (int nb = 0; nb < 4; ++nb) {
      const bf16* bPtr = BT + (size_t)(colBase + nb * 16 + m16) * K + k0 + kh * 16;
      v16bf b = *(const v16bf*)bPtr;             // K = kh*16 + 0..15 contiguous
      acc[nb] = __builtin_amdgcn_wmma_f32_16x16x32_bf16(
          false, a.v, false, b, (short)0, acc[nb], false, false);
    }
  }
#pragma unroll
  for (int nb = 0; nb < 4; ++nb) {
    int col = colBase + nb * 16 + m16;
    float bv = bias ? bias[col] : 0.f;
#pragma unroll
    for (int r = 0; r < 8; ++r) {
      int rowo = rowBase + r + kh * 8;
      float vv = acc[nb][r] + bv;
      if (RELU) vv = fmaxf(vv, 0.f);
      if (OUT_BF16) ((bf16*)Cout)[(size_t)rowo * N + col] = f2bf(vv);
      else          ((float*)Cout)[(size_t)rowo * N + col] = vv;
    }
  }
}

// ---------------- GCN aggregation ----------------
__global__ void deg_kernel(const int* __restrict__ col, float* __restrict__ deg, int E) {
  int e = blockIdx.x * blockDim.x + threadIdx.x;
  if (e >= E) return;
  atomicAdd(&deg[col[e]], 1.0f);
}

__global__ void dis_kernel(const float* __restrict__ deg, float* __restrict__ dis, int n) {
  int i = blockIdx.x * blockDim.x + threadIdx.x;
  if (i >= n) return;
  dis[i] = rsqrtf(deg[i] + 1.0f);   // +1 self loop, always >= 1
}

// one thread = one edge x 8-feature chunk; atomics into agg
__global__ void scatter_kernel(const float* __restrict__ lin, const int* __restrict__ rows,
                               const int* __restrict__ cols, const float* __restrict__ dis,
                               float* __restrict__ agg, int E, int C) {
  int chunks = C >> 3;
  int tid = blockIdx.x * blockDim.x + threadIdx.x;
  if (tid >= E * chunks) return;
  int e  = tid / chunks;
  int c0 = (tid - e * chunks) * 8;
  int r = rows[e], c = cols[e];
  float norm = dis[r] * dis[c];
  const float* src = lin + (size_t)r * C + c0;
  float* dst       = agg + (size_t)c * C + c0;
#pragma unroll
  for (int j = 0; j < 8; ++j) atomicAdd(dst + j, src[j] * norm);
}

// out = [residual +] relu(agg + lin*dis^2 + bias)
__global__ void gcn_finalize_kernel(const float* __restrict__ agg, const float* __restrict__ lin,
                                    const float* __restrict__ dis, const float* __restrict__ bias,
                                    const float* __restrict__ residual, float* __restrict__ out,
                                    int total, int C) {
  int i = blockIdx.x * blockDim.x + threadIdx.x;
  if (i >= total) return;
  int n = i / C, f = i - n * C;
  float dd = dis[n];
  float v = agg[i] + lin[i] * dd * dd + bias[f];
  v = fmaxf(v, 0.f);
  out[i] = residual ? residual[i] + v : v;
}

// t = h + inter_f[batch]; write f32 (in place ok) and bf16
__global__ void add_inter_kernel(const float* __restrict__ h, const float* __restrict__ inter,
                                 const int* __restrict__ batch, float* __restrict__ tf,
                                 bf16* __restrict__ tb, int total, int C) {
  int i = blockIdx.x * blockDim.x + threadIdx.x;
  if (i >= total) return;
  int n = i / C, f = i - n * C;
  float v = h[i] + inter[(size_t)batch[n] * C + f];
  tf[i] = v;
  tb[i] = f2bf(v);
}

// ---------------- fused attention ----------------
// 1 wave per (graph, head, 16-row tile). scores strip in LDS, softmax, P@V.
__global__ void __launch_bounds__(32)
attention_kernel(const bf16* __restrict__ Q, const bf16* __restrict__ Km,
                 const bf16* __restrict__ V, bf16* __restrict__ O) {
  __shared__ __align__(32) float sc[16][SS];
  __shared__ __align__(32) bf16  pb[16][SS];
  const int id = blockIdx.x;
  const int rt = id & 15;
  const int hh = (id >> 4) & 7;
  const int g  = id >> 7;
  const int lane = threadIdx.x;
  const int m16 = lane & 15;
  const int kh  = lane >> 4;

  // Q fragment (16 rows x 32 of this head)
  const bf16* qPtr = Q + (size_t)(g * SS + rt * 16 + m16) * DH + hh * HD + kh * 16;
  AFragU a;
  a.q[0] = *(const uint4a*)qPtr;
  a.q[1] = *(const uint4a*)(qPtr + 8);

  const float scale = 0.1767766953f;   // 1/sqrt(32)
  for (int kt = 0; kt < 16; ++kt) {
    // B fragment = K^T tile: lane n -> key row, contiguous 16 bf16 in d
    const bf16* kPtr = Km + (size_t)(g * SS + kt * 16 + m16) * DH + hh * HD + kh * 16;
    v16bf b = *(const v16bf*)kPtr;
    v8f c = {};
    c = __builtin_amdgcn_wmma_f32_16x16x32_bf16(false, a.v, false, b, (short)0, c, false, false);
#pragma unroll
    for (int r = 0; r < 8; ++r) sc[r + kh * 8][kt * 16 + m16] = c[r] * scale;
  }
  __syncthreads();

  // softmax over each of 16 rows (256 cols), wave-cooperative
  for (int r = 0; r < 16; ++r) {
    float vals[8];
    float mx = -1e30f;
#pragma unroll
    for (int j = 0; j < 8; ++j) { vals[j] = sc[r][lane + j * 32]; mx = fmaxf(mx, vals[j]); }
#pragma unroll
    for (int s = 16; s > 0; s >>= 1) mx = fmaxf(mx, __shfl_xor(mx, s, 32));
    float sum = 0.f;
#pragma unroll
    for (int j = 0; j < 8; ++j) { vals[j] = __expf(vals[j] - mx); sum += vals[j]; }
#pragma unroll
    for (int s = 16; s > 0; s >>= 1) sum += __shfl_xor(sum, s, 32);
    float inv = 1.0f / sum;
#pragma unroll
    for (int j = 0; j < 8; ++j) pb[r][lane + j * 32] = f2bf(vals[j] * inv);
  }
  __syncthreads();

  // O tile (16 x 32) = P(16x256) @ V(256x32), K-loop of 8 x 32
#pragma unroll
  for (int dt = 0; dt < 2; ++dt) {
    v8f c = {};
    for (int ks = 0; ks < 8; ++ks) {
      AFragU pa;
      pa.q[0] = *(const uint4a*)(&pb[m16][ks * 32 + kh * 16]);
      pa.q[1] = *(const uint4a*)(&pb[m16][ks * 32 + kh * 16 + 8]);
      const bf16* vPtr = V + (size_t)(g * SS + ks * 32 + kh * 16) * DH + hh * HD + dt * 16 + m16;
      v16bf b;
#pragma unroll
      for (int j = 0; j < 16; ++j) b[j] = vPtr[(size_t)j * DH];
      c = __builtin_amdgcn_wmma_f32_16x16x32_bf16(false, pa.v, false, b, (short)0, c, false, false);
    }
#pragma unroll
    for (int r = 0; r < 8; ++r)
      O[(size_t)(g * SS + rt * 16 + r + kh * 8) * DH + hh * HD + dt * 16 + m16] = f2bf(c[r]);
  }
}

// ---------------- LayerNorm over rows (C=256), block per row ----------------
__global__ void __launch_bounds__(256)
ln_kernel(const float* __restrict__ A, const float* __restrict__ Badd,
          const float* __restrict__ g, const float* __restrict__ b,
          float* __restrict__ out, bf16* __restrict__ outbf, int C) {
  __shared__ float red[256];
  const int row = blockIdx.x;
  const int f = threadIdx.x;
  float v = A[(size_t)row * C + f] + Badd[(size_t)row * C + f];
  red[f] = v;
  __syncthreads();
  for (int s = 128; s > 0; s >>= 1) { if (f < s) red[f] += red[f + s]; __syncthreads(); }
  float mu = red[0] * (1.0f / 256.0f);
  __syncthreads();
  float d = v - mu;
  red[f] = d * d;
  __syncthreads();
  for (int s = 128; s > 0; s >>= 1) { if (f < s) red[f] += red[f + s]; __syncthreads(); }
  float var = red[0] * (1.0f / 256.0f);
  float y = d * rsqrtf(var + EPS) * g[f] + b[f];
  out[(size_t)row * C + f] = y;
  if (outbf) outbf[(size_t)row * C + f] = f2bf(y);
}

// ---------------- reductions ----------------
__global__ void readout_kernel(const float* __restrict__ t3, float* __restrict__ out) {
  int gph = blockIdx.x, f = threadIdx.x;          // 256 threads
  const float* p = t3 + (size_t)gph * SS * DH + f;
  float s = 0.f;
  for (int r = 0; r < SS; ++r) s += p[(size_t)r * DH];
  out[gph * DH + f] = s;
}

__global__ void init_avg_kernel(const float* __restrict__ x, float* __restrict__ out) {
  int gph = blockIdx.x, f = threadIdx.x;          // 128 threads
  const float* p = x + (size_t)gph * SS * DIN + f;
  float s = 0.f;
  for (int r = 0; r < SS; ++r) s += p[(size_t)r * DIN];
  out[gph * DIN + f] = s * (1.0f / SS);
}

// ---------------- host orchestration ----------------
extern "C" void kernel_launch(void* const* d_in, const int* in_sizes, int n_in,
                              void* d_out, int out_size, void* d_ws, size_t ws_size,
                              hipStream_t stream) {
  (void)in_sizes; (void)n_in; (void)out_size; (void)ws_size;
  const float* x       = (const float*)d_in[0];
  const int*   edges   = (const int*)d_in[1];
  const int*   batch   = (const int*)d_in[2];
  const float* inter_f = (const float*)d_in[3];
  const float* bn1_g = (const float*)d_in[4],  *bn1_b = (const float*)d_in[5];
  const float* bn2_g = (const float*)d_in[6],  *bn2_b = (const float*)d_in[7];
  const float* w_conv1 = (const float*)d_in[8],  *b_conv1 = (const float*)d_in[9];
  const float* w_conv2 = (const float*)d_in[10], *b_conv2 = (const float*)d_in[11];
  const float* wq = (const float*)d_in[12], *bq = (const float*)d_in[13];
  const float* wk = (const float*)d_in[14], *bk = (const float*)d_in[15];
  const float* wv = (const float*)d_in[16], *bv = (const float*)d_in[17];
  const float* wo = (const float*)d_in[18], *bo = (const float*)d_in[19];
  const float* ln1_g = (const float*)d_in[20], *ln1_b = (const float*)d_in[21];
  const float* ln2_g = (const float*)d_in[22], *ln2_b = (const float*)d_in[23];
  const float* w_ff1 = (const float*)d_in[24], *b_ff1 = (const float*)d_in[25];
  const float* w_ff2 = (const float*)d_in[26], *b_ff2 = (const float*)d_in[27];
  float* out = (float*)d_out;
  const int* erow = edges;
  const int* ecol = edges + EE;

  // ---- workspace carve (~235 MB) ----
  char* ws = (char*)d_ws;
  size_t off = 0;
  auto carve = [&](size_t bytes) { char* p = ws + off; off += (bytes + 255) & ~(size_t)255; return p; };
  float* F0   = (float*)carve((size_t)NN * DH * 4);
  float* F1   = (float*)carve((size_t)NN * DH * 4);
  float* F2   = (float*)carve((size_t)NN * DH * 4);
  bf16*  BF0  = (bf16*) carve((size_t)NN * DH * 2);
  bf16*  BF1  = (bf16*) carve((size_t)NN * DH * 2);
  bf16*  BF2  = (bf16*) carve((size_t)NN * DH * 2);
  bf16*  BF3  = (bf16*) carve((size_t)NN * DH * 2);
  bf16*  FFB  = (bf16*) carve((size_t)NN * DFF * 2);
  bf16*  XBN  = (bf16*) carve((size_t)NN * DIN * 2);
  bf16*  w1T  = (bf16*) carve((size_t)DIN * DH * 2);
  bf16*  w2T  = (bf16*) carve((size_t)DH * DH * 2);
  bf16*  wqT  = (bf16*) carve((size_t)DH * DH * 2);
  bf16*  wkT  = (bf16*) carve((size_t)DH * DH * 2);
  bf16*  wvT  = (bf16*) carve((size_t)DH * DH * 2);
  bf16*  woT  = (bf16*) carve((size_t)DH * DH * 2);
  bf16*  ff1T = (bf16*) carve((size_t)DH * DFF * 2);
  bf16*  ff2T = (bf16*) carve((size_t)DFF * DH * 2);
  float* deg  = (float*)carve((size_t)NN * 4);
  float* dis  = (float*)carve((size_t)NN * 4);
  float* sums = (float*)carve(256 * 4);
  float* sumq = (float*)carve(256 * 4);
  float* mean = (float*)carve(256 * 4);
  float* istd = (float*)carve(256 * 4);

  // ---- independent output: init_avg_h ----
  init_avg_kernel<<<BB, DIN, 0, stream>>>(x, out + (size_t)BB * DH);

  // ---- weight convert+transpose ----
  wconv_kernel<<<(DIN * DH + 255) / 256, 256, 0, stream>>>(w_conv1, w1T, DIN, DH);
  wconv_kernel<<<(DH * DH + 255) / 256, 256, 0, stream>>>(w_conv2, w2T, DH, DH);
  wconv_kernel<<<(DH * DH + 255) / 256, 256, 0, stream>>>(wq, wqT, DH, DH);
  wconv_kernel<<<(DH * DH + 255) / 256, 256, 0, stream>>>(wk, wkT, DH, DH);
  wconv_kernel<<<(DH * DH + 255) / 256, 256, 0, stream>>>(wv, wvT, DH, DH);
  wconv_kernel<<<(DH * DH + 255) / 256, 256, 0, stream>>>(wo, woT, DH, DH);
  wconv_kernel<<<(DH * DFF + 255) / 256, 256, 0, stream>>>(w_ff1, ff1T, DH, DFF);
  wconv_kernel<<<(DFF * DH + 255) / 256, 256, 0, stream>>>(w_ff2, ff2T, DFF, DH);

  // ---- degree norms ----
  hipMemsetAsync(deg, 0, (size_t)NN * 4, stream);
  deg_kernel<<<(EE + 255) / 256, 256, 0, stream>>>(ecol, deg, EE);
  dis_kernel<<<(NN + 255) / 256, 256, 0, stream>>>(deg, dis, NN);

  // ---- BN1 on x -> XBN(bf16) ----
  hipMemsetAsync(sums, 0, 256 * 4, stream);
  hipMemsetAsync(sumq, 0, 256 * 4, stream);
  colsum_kernel<<<NN / 256, DIN, 0, stream>>>(x, sums, sumq, DIN, 256);
  bnstats_kernel<<<1, DIN, 0, stream>>>(sums, sumq, mean, istd, DIN, 1.0f / NN);
  bn_apply_kernel<<<(NN * DIN + 255) / 256, 256, 0, stream>>>(x, mean, istd, bn1_g, bn1_b,
                                                              XBN, NN * DIN, DIN);

  // ---- conv1 linear: F0 = XBN @ W1 ----
  gemm_bf16_kernel<false, false><<<dim3(DH / 64, NN / 128), 256, 0, stream>>>(
      XBN, w1T, nullptr, F0, NN, DH, DIN);
  // aggregate + finalize -> h1 in F1
  hipMemsetAsync(F1, 0, (size_t)NN * DH * 4, stream);
  scatter_kernel<<<(EE * (DH / 8) + 255) / 256, 256, 0, stream>>>(F0, erow, ecol, dis, F1, EE, DH);
  gcn_finalize_kernel<<<(NN * DH + 255) / 256, 256, 0, stream>>>(F1, F0, dis, b_conv1, nullptr,
                                                                  F1, NN * DH, DH);

  // ---- BN2 on h1 -> BF0(bf16) ----
  hipMemsetAsync(sums, 0, 256 * 4, stream);
  hipMemsetAsync(sumq, 0, 256 * 4, stream);
  colsum_kernel<<<NN / 256, DH, 0, stream>>>(F1, sums, sumq, DH, 256);
  bnstats_kernel<<<1, DH, 0, stream>>>(sums, sumq, mean, istd, DH, 1.0f / NN);
  bn_apply_kernel<<<(NN * DH + 255) / 256, 256, 0, stream>>>(F1, mean, istd, bn2_g, bn2_b,
                                                              BF0, NN * DH, DH);

  // ---- conv2 linear: F0 = BF0 @ W2; aggregate into F2; h = h1 + relu(...) in F2 ----
  gemm_bf16_kernel<false, false><<<dim3(DH / 64, NN / 128), 256, 0, stream>>>(
      BF0, w2T, nullptr, F0, NN, DH, DH);
  hipMemsetAsync(F2, 0, (size_t)NN * DH * 4, stream);
  scatter_kernel<<<(EE * (DH / 8) + 255) / 256, 256, 0, stream>>>(F0, erow, ecol, dis, F2, EE, DH);
  gcn_finalize_kernel<<<(NN * DH + 255) / 256, 256, 0, stream>>>(F2, F0, dis, b_conv2, F1,
                                                                  F2, NN * DH, DH);

  // ---- t = h + inter_f[batch] : f32 in F2 (in place), bf16 in BF0 ----
  add_inter_kernel<<<(NN * DH + 255) / 256, 256, 0, stream>>>(F2, inter_f, batch, F2, BF0,
                                                               NN * DH, DH);

  // ---- Q, K, V projections (bf16 out) ----
  gemm_bf16_kernel<true, false><<<dim3(DH / 64, NN / 128), 256, 0, stream>>>(
      BF0, wqT, bq, BF1, NN, DH, DH);
  gemm_bf16_kernel<true, false><<<dim3(DH / 64, NN / 128), 256, 0, stream>>>(
      BF0, wkT, bk, BF2, NN, DH, DH);
  gemm_bf16_kernel<true, false><<<dim3(DH / 64, NN / 128), 256, 0, stream>>>(
      BF0, wvT, bv, BF3, NN, DH, DH);

  // ---- fused attention: O(bf16) into BF0 ----
  attention_kernel<<<BB * HH * (SS / 16), 32, 0, stream>>>(BF1, BF2, BF3, BF0);

  // ---- output projection: F0 = O @ Wo + bo ----
  gemm_bf16_kernel<false, false><<<dim3(DH / 64, NN / 128), 256, 0, stream>>>(
      BF0, woT, bo, F0, NN, DH, DH);

  // ---- t2 = LN(t + oproj): f32 in F1, bf16 in BF1 ----
  ln_kernel<<<NN, DH, 0, stream>>>(F2, F0, ln1_g, ln1_b, F1, BF1, DH);

  // ---- FFN: FFB = relu(t2 @ Wff1 + b1) (bf16); F0 = FFB @ Wff2 + b2 ----
  gemm_bf16_kernel<true, true><<<dim3(DFF / 64, NN / 128), 256, 0, stream>>>(
      BF1, ff1T, b_ff1, FFB, NN, DFF, DH);
  gemm_bf16_kernel<false, false><<<dim3(DH / 64, NN / 128), 256, 0, stream>>>(
      FFB, ff2T, b_ff2, F0, NN, DH, DFF);

  // ---- t3 = LN(t2 + ffn): f32 in F2 ----
  ln_kernel<<<NN, DH, 0, stream>>>(F1, F0, ln2_g, ln2_b, F2, nullptr, DH);

  // ---- readout ----
  readout_kernel<<<BB, DH, 0, stream>>>(F2, out);
}
